// SharedAttentionProcessor_18528488915095
// MI455X (gfx1250) — matrix operand order repaired and checked by reading, not compiled, not run on hardware
//
#include <hip/hip_runtime.h>

// SharedAttentionProcessor for MI455X (gfx1250, wave32, WMMA + async-to-LDS).
// B=4, S=1024, C=512, H=8, D=64.  Matrix math via v_wmma_f32_16x16x32_f16;
// attention K/V staging via GLOBAL_LOAD_ASYNC_TO_LDS_B128 (ASYNCcnt).

#define B_ 4
#define S_ 1024
#define C_ 512
#define H_ 8
#define D_ 64

static constexpr float SHIFT_ = 0.6931471805599453f; // ln(2) bias on shared keys
static constexpr float EPS_   = 1e-5f;
static constexpr float SCALE_ = 0.125f;              // 1/sqrt(D)

typedef __attribute__((ext_vector_type(16))) _Float16 v16h;
typedef __attribute__((ext_vector_type(8)))  float    v8f;

static __device__ __forceinline__ v8f wmma_f16(v16h a, v16h b, v8f c) {
  // (neg_a, A, neg_b, B, c_mod, C, reuse_a, reuse_b)
  return __builtin_amdgcn_wmma_f32_16x16x32_f16(false, a, false, b, (short)0, c,
                                                false, false);
}

// A-matrix (16x32 f16) element j -> K index, per documented CDNA5 layout.
static __device__ __forceinline__ int a_k_of(int j, int laneHi) {
  return (j < 8 ? j : j + 8) + (laneHi ? 8 : 0);
}

// CDNA5 async memory->LDS copy, 16B per lane, tracked by ASYNCcnt.
static __device__ __forceinline__ void async_b128(_Float16* lds,
                                                  const _Float16* g) {
  const unsigned int       l = (unsigned int)(uintptr_t)lds;  // LDS byte offset
  const unsigned long long a = (unsigned long long)(uintptr_t)g;
  asm volatile("global_load_async_to_lds_b128 %0, %1, off"
               :
               : "v"(l), "v"(a)
               : "memory");
}
static __device__ __forceinline__ void wait_async0() {
  asm volatile("s_wait_asynccnt 0" ::: "memory");
}

// ---------------------------------------------------------------------------
// Kernel 1: Q/K/V projections.  out[b,h,s,d] = sum_c x[b,s,c] * W[o,c], o=h*64+d
// Per-wave 16x64 tile, K=512 in 16 WMMA steps, f16 inputs / f32 accumulate.
// ---------------------------------------------------------------------------
__global__ __launch_bounds__(256) void qkv_proj_kernel(
    const float* __restrict__ x,
    const float* __restrict__ Wq, const float* __restrict__ Wk,
    const float* __restrict__ Wv,
    float* __restrict__ q32, float* __restrict__ k32, float* __restrict__ v32) {
  const int wave   = (blockIdx.x * blockDim.x + threadIdx.x) >> 5;
  const int lane   = threadIdx.x & 31;
  const int laneHi = lane >> 4;
  const int ln     = lane & 15;

  const int mat = wave / (256 * 8);     // 0=Q 1=K 2=V
  const int rem = wave % (256 * 8);
  const int mt  = rem >> 3;             // 256 row tiles of 16
  const int ng  = rem & 7;              // 8 col groups of 64

  const float* W  = (mat == 0) ? Wq : (mat == 1) ? Wk : Wv;
  float* out      = (mat == 0) ? q32 : (mat == 1) ? k32 : v32;

  const int rowA = mt * 16 + ln;

  v8f acc[4];
#pragma unroll
  for (int t = 0; t < 4; ++t)
#pragma unroll
    for (int i = 0; i < 8; ++i) acc[t][i] = 0.f;

  for (int s = 0; s < 16; ++s) {
    v16h a;
    {
      const int c0 = 32 * s + (laneHi ? 16 : 0);
      const float* xp = x + rowA * C_ + c0;
#pragma unroll
      for (int j = 0; j < 8; ++j) a[j] = (_Float16)xp[j];
#pragma unroll
      for (int j = 0; j < 8; ++j) a[8 + j] = (_Float16)xp[8 + j];
    }
#pragma unroll
    for (int t = 0; t < 4; ++t) {
      const int o  = ng * 64 + t * 16 + ln;
      const int c0 = 32 * s + (laneHi ? 16 : 0);
      const float* wp = W + o * C_ + c0;
      v16h bf;
#pragma unroll
      for (int j = 0; j < 16; ++j) bf[j] = (_Float16)wp[j];
      acc[t] = wmma_f16(a, bf, acc[t]);
    }
  }
#pragma unroll
  for (int t = 0; t < 4; ++t) {
    const int o = ng * 64 + t * 16 + ln;
    const int h = o >> 6, d = o & 63;
#pragma unroll
    for (int i = 0; i < 8; ++i) {
      const int m = mt * 16 + i + 8 * laneHi;
      const int b = m >> 10, seq = m & 1023;
      out[((b * H_ + h) * S_ + seq) * D_ + d] = acc[t][i];
    }
  }
}

// ---------------------------------------------------------------------------
// Kernel 2: per-(b,h,d) mean and sqrt(var_unbiased + eps) over the S axis.
// ---------------------------------------------------------------------------
__global__ __launch_bounds__(256) void stats_kernel(
    const float* __restrict__ q32, const float* __restrict__ k32,
    float* __restrict__ meanq, float* __restrict__ stdq,
    float* __restrict__ meank, float* __restrict__ stdk) {
  const int bh = blockIdx.x;                       // 0..31
  const float* src = (blockIdx.y == 0) ? q32 : k32;
  float* mdst = (blockIdx.y == 0) ? meanq : meank;
  float* sdst = (blockIdx.y == 0) ? stdq : stdk;

  const int d  = threadIdx.x & 63;
  const int sg = threadIdx.x >> 6;                 // 0..3
  const float* p = src + (size_t)bh * S_ * D_;

  float sum = 0.f, sq = 0.f;
  for (int s = sg; s < S_; s += 4) {
    const float v = p[s * D_ + d];
    sum += v;
    sq  += v * v;
  }
  __shared__ float ls1[4][64];
  __shared__ float ls2[4][64];
  ls1[sg][d] = sum;
  ls2[sg][d] = sq;
  __syncthreads();
  if (sg == 0) {
    const float s1 = ls1[0][d] + ls1[1][d] + ls1[2][d] + ls1[3][d];
    const float s2 = ls2[0][d] + ls2[1][d] + ls2[2][d] + ls2[3][d];
    const float mean = s1 * (1.0f / S_);
    const float var  = (s2 - (float)S_ * mean * mean) * (1.0f / (S_ - 1));
    mdst[bh * D_ + d] = mean;
    sdst[bh * D_ + d] = sqrtf(var + EPS_);
  }
}

// ---------------------------------------------------------------------------
// Kernel 3: pack K/V for the attention loop (done ONCE instead of once per
// query-block).  kpack[bh][j][d] f16, j in [0,2S): AdaIN'd keys then raw
// style-source keys (AdaIN is identity on the style batches).
// vpack[bh][blk][d][key] f16: 32-key blocks pre-transposed for PV B-fragments,
// with channel d=0 nulled.
// ---------------------------------------------------------------------------
__global__ __launch_bounds__(256) void pack_kernel(
    const float* __restrict__ k32, const float* __restrict__ v32,
    const float* __restrict__ meank, const float* __restrict__ stdk,
    _Float16* __restrict__ kpack, _Float16* __restrict__ vpack) {
  const int blk = blockIdx.x;           // 0..63 key blocks of 32 over 2S
  const int h = blockIdx.y, b = blockIdx.z;
  const int sb = (b < B_ / 2) ? 0 : (B_ / 2);
  const int bh = b * H_ + h, sbh = sb * H_ + h;
  const int tid = threadIdx.x;
  const int key = tid >> 3;             // 0..31
  const int d0  = (tid & 7) * 8;
  const int j   = blk * 32 + key;
  const bool second = (j >= S_);
  const int sh = second ? sbh : bh;
  const int r  = second ? (j - S_) : j;

  const float* kp = k32 + ((size_t)sh * S_ + r) * D_ + d0;
  const float* vp = v32 + ((size_t)sh * S_ + r) * D_ + d0;
  _Float16* kout = kpack + ((size_t)bh * (2 * S_) + j) * D_ + d0;
  _Float16* vout = vpack + ((size_t)bh * 64 + blk) * (D_ * 32);

#pragma unroll
  for (int jj = 0; jj < 8; ++jj) {
    const int d = d0 + jj;
    float kv = kp[jj];
    if (!second) {
      const float a = stdk[sbh * D_ + d] / stdk[bh * D_ + d];
      const float c = meank[sbh * D_ + d] - meank[bh * D_ + d] * a;
      kv = kv * a + c;
    }
    kout[jj] = (_Float16)kv;
    float vv = vp[jj];
    if (d == 0) vv = 0.f;               // channel_to_null
    vout[d * 32 + key] = (_Float16)vv;
  }
}

// ---------------------------------------------------------------------------
// Kernel 4: flash attention over 2S keys.
// Workgroup = 8 waves x 16 query rows = 128 queries; grid (S/128, H, B).
// K/V tiles arrive via double-buffered GLOBAL_LOAD_ASYNC_TO_LDS_B128;
// consumption gated by s_wait_asynccnt 0 + workgroup barrier.
// ---------------------------------------------------------------------------
__global__ __launch_bounds__(256) void attn_kernel(
    const float* __restrict__ q32,
    const float* __restrict__ meanq, const float* __restrict__ stdq,
    const _Float16* __restrict__ kpack, const _Float16* __restrict__ vpack,
    _Float16* __restrict__ attn_out) {
  const int qblk = blockIdx.x;
  const int h    = blockIdx.y;
  const int b    = blockIdx.z;
  const int sb   = (b < B_ / 2) ? 0 : (B_ / 2);
  const int tid  = threadIdx.x;
  const int wave = tid >> 5, lane = tid & 31;
  const int laneHi = lane >> 4, ln = lane & 15;

  const int bh  = b * H_ + h;
  const int sbh = sb * H_ + h;

  __shared__ __align__(16) _Float16 Kt[2][32][64];  // [buf][key][d]
  __shared__ __align__(16) _Float16 Vt[2][64][32];  // [buf][d][key]
  __shared__ __align__(16) _Float16 Pb[8][16][32];  // per-wave P transpose

  const _Float16* kbase = kpack + (size_t)bh * (2 * S_) * D_;
  const _Float16* vbase = vpack + (size_t)bh * 64 * (D_ * 32);

  // Prologue: async-stage key block 0 into buffer 0 (16B per thread per tile).
  async_b128(&Kt[0][0][0] + tid * 8, kbase + tid * 8);
  async_b128(&Vt[0][0][0] + tid * 8, vbase + tid * 8);

  // Q A-fragments for this wave's 16 rows, AdaIN applied on the fly.
  const int q0 = qblk * 128 + wave * 16;
  v16h Qa[2];
  {
    const int m = q0 + ln;
    const float* qp = q32 + ((size_t)bh * S_ + m) * D_;
#pragma unroll
    for (int s = 0; s < 2; ++s)
#pragma unroll
      for (int j = 0; j < 16; ++j) {
        const int d = 32 * s + a_k_of(j, laneHi);
        const float a = stdq[sbh * D_ + d] / stdq[bh * D_ + d];
        const float c = meanq[sbh * D_ + d] - meanq[bh * D_ + d] * a;
        Qa[s][j] = (_Float16)(qp[d] * a + c);
      }
  }

  v8f O[4];
  float mrow[8], lrow[8];
#pragma unroll
  for (int t = 0; t < 4; ++t)
#pragma unroll
    for (int i = 0; i < 8; ++i) O[t][i] = 0.f;
#pragma unroll
  for (int i = 0; i < 8; ++i) { mrow[i] = -1e30f; lrow[i] = 0.f; }

  for (int t = 0; t < (2 * S_) / 32; ++t) {        // 64 key blocks of 32
    const int cb = t & 1;

    // All waves drain their own async copies into buffer cb, then rendezvous.
    wait_async0();
    __syncthreads();

    // Kick off next block into the other buffer (safe: everyone finished
    // reading it during iteration t-1, enforced by the barrier above).
    if (t + 1 < (2 * S_) / 32) {
      const int nb = (t + 1) & 1;
      async_b128(&Kt[nb][0][0] + tid * 8,
                 kbase + (size_t)(t + 1) * 32 * D_ + tid * 8);
      async_b128(&Vt[nb][0][0] + tid * 8,
                 vbase + (size_t)(t + 1) * (D_ * 32) + tid * 8);
    }

    const bool second = (t * 32 >= S_);

    // Q @ K^T : two 16-key column tiles, K=64 in two WMMA steps each.
    v8f L[2];
#pragma unroll
    for (int g = 0; g < 2; ++g) {
      v8f c;
#pragma unroll
      for (int i = 0; i < 8; ++i) c[i] = 0.f;
#pragma unroll
      for (int s = 0; s < 2; ++s) {
        const int keyLocal = g * 16 + ln;
        const int dBase    = 32 * s + (laneHi ? 8 : 0);
        const _Float16* src = &Kt[cb][keyLocal][dBase];
        v16h bf;
#pragma unroll
        for (int j = 0; j < 16; ++j) bf[j] = src[j < 8 ? j : j + 8];
        c = wmma_f16(Qa[s], bf, c);
      }
      L[g] = c;
    }

    const float add = second ? SHIFT_ : 0.f;
#pragma unroll
    for (int g = 0; g < 2; ++g)
#pragma unroll
      for (int i = 0; i < 8; ++i) L[g][i] = L[g][i] * SCALE_ + add;

    // Row max over the 32 block columns (xor-shuffle across the 16-lane half).
    float bm[8];
#pragma unroll
    for (int i = 0; i < 8; ++i) bm[i] = fmaxf(L[0][i], L[1][i]);
#pragma unroll
    for (int mask = 1; mask < 16; mask <<= 1)
#pragma unroll
      for (int i = 0; i < 8; ++i)
        bm[i] = fmaxf(bm[i], __shfl_xor(bm[i], mask, 32));

    float rs[8];
#pragma unroll
    for (int i = 0; i < 8; ++i) {
      const float mn   = fmaxf(mrow[i], bm[i]);
      const float corr = __expf(mrow[i] - mn);
      mrow[i] = mn;
      lrow[i] *= corr;
#pragma unroll
      for (int tt = 0; tt < 4; ++tt) O[tt][i] *= corr;
      const float e0 = __expf(L[0][i] - mn);
      const float e1 = __expf(L[1][i] - mn);
      L[0][i] = e0;
      L[1][i] = e1;
      rs[i]   = e0 + e1;
    }
#pragma unroll
    for (int mask = 1; mask < 16; mask <<= 1)
#pragma unroll
      for (int i = 0; i < 8; ++i) rs[i] += __shfl_xor(rs[i], mask, 32);
#pragma unroll
    for (int i = 0; i < 8; ++i) lrow[i] += rs[i];

    // P (C-layout, per-lane columns) -> LDS -> A-fragment (per-lane rows).
#pragma unroll
    for (int g = 0; g < 2; ++g)
#pragma unroll
      for (int i = 0; i < 8; ++i)
        Pb[wave][i + 8 * laneHi][g * 16 + ln] = (_Float16)L[g][i];

    v16h Pa;
    {
      const _Float16* pr = &Pb[wave][ln][laneHi ? 16 : 0];
#pragma unroll
      for (int j = 0; j < 8; ++j) Pa[j] = pr[j];
#pragma unroll
      for (int j = 0; j < 8; ++j) Pa[8 + j] = pr[8 + j];
    }

    // P @ V : 4 output column tiles of 16 over D=64, K=32 keys.
#pragma unroll
    for (int tt = 0; tt < 4; ++tt) {
      const int d = tt * 16 + ln;
      const _Float16* vsrc = &Vt[cb][d][laneHi ? 16 : 0];
      v16h bf;
#pragma unroll
      for (int j = 0; j < 16; ++j) bf[j] = vsrc[j];
      O[tt] = wmma_f16(Pa, bf, O[tt]);
    }
  }

  // Epilogue: normalize and store merged-head f16 [B,S,C].
#pragma unroll
  for (int tt = 0; tt < 4; ++tt)
#pragma unroll
    for (int i = 0; i < 8; ++i) {
      const int m = q0 + i + 8 * laneHi;
      const int d = tt * 16 + ln;
      attn_out[((size_t)b * S_ + m) * C_ + h * D_ + d] =
          (_Float16)(O[tt][i] / lrow[i]);
    }
}

// ---------------------------------------------------------------------------
// Kernel 5: output projection.  out[m,o] = sum_c ao[m,c] * Wo[o,c]  (f32 out)
// ---------------------------------------------------------------------------
__global__ __launch_bounds__(256) void out_proj_kernel(
    const _Float16* __restrict__ ao, const float* __restrict__ Wo,
    float* __restrict__ out) {
  const int wave   = (blockIdx.x * blockDim.x + threadIdx.x) >> 5;
  const int lane   = threadIdx.x & 31;
  const int laneHi = lane >> 4;
  const int ln     = lane & 15;
  const int mt = wave >> 3;
  const int ng = wave & 7;
  const int rowA = mt * 16 + ln;

  v8f acc[4];
#pragma unroll
  for (int t = 0; t < 4; ++t)
#pragma unroll
    for (int i = 0; i < 8; ++i) acc[t][i] = 0.f;

  for (int s = 0; s < 16; ++s) {
    v16h a;
    {
      const int c0 = 32 * s + (laneHi ? 16 : 0);
      const _Float16* ap = ao + (size_t)rowA * C_ + c0;
#pragma unroll
      for (int j = 0; j < 8; ++j) a[j] = ap[j];
#pragma unroll
      for (int j = 0; j < 8; ++j) a[8 + j] = ap[8 + j];
    }
#pragma unroll
    for (int t = 0; t < 4; ++t) {
      const int o  = ng * 64 + t * 16 + ln;
      const int c0 = 32 * s + (laneHi ? 16 : 0);
      const float* wp = Wo + o * C_ + c0;
      v16h bf;
#pragma unroll
      for (int j = 0; j < 16; ++j) bf[j] = (_Float16)wp[j];
      acc[t] = wmma_f16(a, bf, acc[t]);
    }
  }
#pragma unroll
  for (int t = 0; t < 4; ++t) {
    const int o = ng * 64 + t * 16 + ln;
#pragma unroll
    for (int i = 0; i < 8; ++i) {
      const int m = mt * 16 + i + 8 * laneHi;
      out[(size_t)m * C_ + o] = acc[t][i];
    }
  }
}

// ---------------------------------------------------------------------------
extern "C" void kernel_launch(void* const* d_in, const int* in_sizes, int n_in,
                              void* d_out, int out_size, void* d_ws,
                              size_t ws_size, hipStream_t stream) {
  (void)in_sizes; (void)n_in; (void)out_size; (void)ws_size;
  const float* x  = (const float*)d_in[0];
  const float* Wq = (const float*)d_in[1];
  const float* Wk = (const float*)d_in[2];
  const float* Wv = (const float*)d_in[3];
  const float* Wo = (const float*)d_in[4];
  float* out = (float*)d_out;

  // Workspace layout (~44 MB; entire working set is L2-resident on MI455X):
  float* q32   = (float*)d_ws;                       // [B,H,S,D] f32 (8MB)
  float* k32   = q32 + (size_t)B_ * S_ * C_;         // 8MB
  float* v32   = k32 + (size_t)B_ * S_ * C_;         // 8MB
  float* meanq = v32 + (size_t)B_ * S_ * C_;         // [B,H,D]
  float* stdq  = meanq + B_ * H_ * D_;
  float* meank = stdq + B_ * H_ * D_;
  float* stdk  = meank + B_ * H_ * D_;
  _Float16* kpack = (_Float16*)(stdk + B_ * H_ * D_);        // [B,H,2S,D] (8MB)
  _Float16* vpack = kpack + (size_t)B_ * H_ * 2 * S_ * D_;   // [B,H,64,D,32]
  _Float16* ao    = vpack + (size_t)B_ * H_ * 2 * S_ * D_;   // [B,S,C] f16

  // 1) QKV projections: 3 * 2048 wave-tiles, 8 waves per 256-thread block.
  qkv_proj_kernel<<<768, 256, 0, stream>>>(x, Wq, Wk, Wv, q32, k32, v32);
  // 2) AdaIN statistics over S for Q and K.
  stats_kernel<<<dim3(32, 2), 256, 0, stream>>>(q32, k32, meanq, stdq, meank,
                                                stdk);
  // 3) One-shot K/V pack: AdaIN affine + concat + V-null + f16 + V transpose.
  pack_kernel<<<dim3((2 * S_) / 32, H_, B_), 256, 0, stream>>>(
      k32, v32, meank, stdk, kpack, vpack);
  // 4) Flash attention with async double-buffered LDS staging.
  attn_kernel<<<dim3(S_ / 128, H_, B_), 256, 0, stream>>>(
      q32, meanq, stdq, kpack, vpack, ao);
  // 5) Output projection to f32.
  out_proj_kernel<<<256, 256, 0, stream>>>(ao, Wo, out);
}
